// dctmodule2D_27315992003321
// MI455X (gfx1250) — matrix operand order repaired and checked by reading, not compiled, not run on hardware
//
#include <hip/hip_runtime.h>
#include <stdint.h>

#define NDIM 512
#define NSLICE 96
#define BM 128
#define BN 128
#define BK 32
#define WM 64
#define WN 32
#define MF 4   // WM/16
#define NF 2   // WN/16
#define NTHREADS 256
#define KITERS (NDIM / BK)

typedef __attribute__((ext_vector_type(16))) __bf16 v16bf;
typedef __attribute__((ext_vector_type(8)))  float  v8f;

// ---------------------------------------------------------------------------
// CDNA5 async global->LDS copy (ASYNCcnt path) + wait
// ---------------------------------------------------------------------------
__device__ __forceinline__ uint32_t lds_off(const void* p) {
  // LDS aperture: low 32 bits of a generic shared-pointer are the LDS address
  return (uint32_t)(uintptr_t)p;
}

__device__ __forceinline__ void async_copy_b128(uint32_t lds, uint64_t gaddr) {
  asm volatile("global_load_async_to_lds_b128 %0, %1, off"
               :: "v"(lds), "v"(gaddr) : "memory");
}

__device__ __forceinline__ void wait_async0() {
  asm volatile("s_wait_asynccnt 0" ::: "memory");
}

// ---------------------------------------------------------------------------
// fp32 -> (bf16 hi, bf16 lo) error-compensated split
// ---------------------------------------------------------------------------
__device__ __forceinline__ void split_bf16(float x, uint16_t& h, uint16_t& l) {
  uint32_t u = __float_as_uint(x);
  h = (uint16_t)(u >> 16);                         // truncated hi
  float fh = __uint_as_float(u & 0xffff0000u);
  uint32_t ur = __float_as_uint(x - fh);           // exact residual
  l = (uint16_t)((ur + 0x8000u) >> 16);            // RN lo
}

// ---------------------------------------------------------------------------
// WMMA fragment loads (layouts per CDNA5 ISA 7.12.2, 16-bit A 16x32 / B 32x16)
// ---------------------------------------------------------------------------
__device__ __forceinline__ v16bf ldsfragA(const uint16_t* t, int row, int half) {
  union { uint4 u[2]; v16bf v; } r;
  r.u[0] = *(const uint4*)(t + row * BK + 8 * half);        // K 0..7  (+8)
  r.u[1] = *(const uint4*)(t + row * BK + 16 + 8 * half);   // K 16..23(+8)
  return r.v;
}

__device__ __forceinline__ v16bf ldsfragB(const uint16_t* t, int row, int half) {
  union { uint4 u[2]; v16bf v; } r;
  r.u[0] = *(const uint4*)(t + row * BK + 8 * half);        // K 0..7  (+8)
  r.u[1] = *(const uint4*)(t + row * BK + 16 + 8 * half);   // K 16..23(+8)
  return r.v;
}

__device__ __forceinline__ v8f wmma_bf16(v16bf a, v16bf b, v8f c) {
  return __builtin_amdgcn_wmma_f32_16x16x32_bf16(
      /*neg_a=*/false, a, /*neg_b=*/false, b,
      /*c_mod=*/(short)0, c, /*reuse_a=*/false, /*reuse_b=*/false);
}

// ---------------------------------------------------------------------------
// Build ortho DCT-II matrix C[k][n] = s_k * cos(pi*k*(2n+1)/(2N)), bf16 split
// ---------------------------------------------------------------------------
__global__ void dct_gen_mat(uint16_t* __restrict__ Gh, uint16_t* __restrict__ Gl) {
  int n = blockIdx.x * blockDim.x + threadIdx.x;
  int k = blockIdx.y;
  if (n >= NDIM || k >= NDIM) return;
  int ph = (k * (2 * n + 1)) & (4 * NDIM - 1);          // exact phase mod 4N
  float ang = (float)ph * 3.0679615757712823e-3f;       // * pi/1024
  float sc = (k == 0) ? 4.4194173824159216e-2f          // sqrt(1/512)
                      : 6.25e-2f;                       // sqrt(2/512)
  uint16_t hh, ll;
  split_bf16(sc * cosf(ang), hh, ll);
  Gh[k * NDIM + n] = hh;
  Gl[k * NDIM + n] = ll;
}

// ---------------------------------------------------------------------------
// Pass 1: T^t[s][k1][h] = sum_n X[s][h][n] * C[k1][n]   (stored split bf16)
// Double-buffered: async DMA for tile i+1 overlaps WMMA on tile i.
// ---------------------------------------------------------------------------
__global__ __launch_bounds__(NTHREADS) void dct_pass1(
    const float* __restrict__ X,
    const uint16_t* __restrict__ Gh, const uint16_t* __restrict__ Gl,
    uint16_t* __restrict__ Th, uint16_t* __restrict__ Tl)
{
  __shared__ __align__(16) uint16_t sAh[2][BM * BK];
  __shared__ __align__(16) uint16_t sAl[2][BM * BK];
  __shared__ __align__(16) uint16_t sBh[2][BN * BK];
  __shared__ __align__(16) uint16_t sBl[2][BN * BK];

  const int s    = blockIdx.z;
  const int bm   = blockIdx.y * BM;
  const int bn   = blockIdx.x * BN;
  const int tid  = threadIdx.x;
  const int lane = tid & 31;
  const int wave = tid >> 5;
  const int half = lane >> 4;
  const int l15  = lane & 15;
  const int wm   = (wave >> 2) * WM;
  const int wn   = (wave & 3) * WN;

  const float* Xs = X + (size_t)s * NDIM * NDIM;

  // staging helper (A: fp32 load + split + LDS store; B: async DMA)
  auto stage = [&](int k0, int buf) {
#pragma unroll
    for (int it = 0; it < (BN * BK / 8) / NTHREADS; ++it) {
      int i = tid + it * NTHREADS;
      int r = i >> 2;
      int c = (i & 3) << 3;
      size_t g = (size_t)(bn + r) * NDIM + (k0 + c);
      async_copy_b128(lds_off(&sBh[buf][r * BK + c]),
                      (uint64_t)(uintptr_t)(Gh + g));
      async_copy_b128(lds_off(&sBl[buf][r * BK + c]),
                      (uint64_t)(uintptr_t)(Gl + g));
    }
#pragma unroll
    for (int it = 0; it < (BM * BK / 4) / NTHREADS; ++it) {
      int i = tid + it * NTHREADS;
      int r = i >> 3;
      int c = (i & 7) << 2;
      float4 x4 = *(const float4*)(Xs + (size_t)(bm + r) * NDIM + k0 + c);
      uint16_t h0, l0, h1, l1, h2, l2, h3, l3;
      split_bf16(x4.x, h0, l0); split_bf16(x4.y, h1, l1);
      split_bf16(x4.z, h2, l2); split_bf16(x4.w, h3, l3);
      *(ushort4*)&sAh[buf][r * BK + c] = make_ushort4(h0, h1, h2, h3);
      *(ushort4*)&sAl[buf][r * BK + c] = make_ushort4(l0, l1, l2, l3);
    }
  };

  v8f acc[MF][NF];
#pragma unroll
  for (int i = 0; i < MF; ++i)
#pragma unroll
    for (int j = 0; j < NF; ++j) acc[i][j] = (v8f)(0.0f);

  // prologue: fill buffer 0
  stage(0, 0);
  wait_async0();
  __syncthreads();

#pragma unroll 1
  for (int ki = 0; ki < KITERS; ++ki) {
    const int buf  = ki & 1;
    const int nbuf = buf ^ 1;

    // kick DMA + staging for next tile (overlaps with WMMA below)
    if (ki + 1 < KITERS) stage((ki + 1) * BK, nbuf);

    const uint16_t* cAh = sAh[buf];
    const uint16_t* cAl = sAl[buf];
    const uint16_t* cBh = sBh[buf];
    const uint16_t* cBl = sBl[buf];

    v16bf ah[MF], al[MF], bh[NF], bl[NF];
#pragma unroll
    for (int mf = 0; mf < MF; ++mf) {
      int row = wm + mf * 16 + l15;
      ah[mf] = ldsfragA(cAh, row, half);
      al[mf] = ldsfragA(cAl, row, half);
    }
#pragma unroll
    for (int nf = 0; nf < NF; ++nf) {
      int row = wn + nf * 16 + l15;
      bh[nf] = ldsfragB(cBh, row, half);
      bl[nf] = ldsfragB(cBl, row, half);
    }
#pragma unroll
    for (int mf = 0; mf < MF; ++mf)
#pragma unroll
      for (int nf = 0; nf < NF; ++nf) {
        acc[mf][nf] = wmma_bf16(ah[mf], bh[nf], acc[mf][nf]);  // hi*hi
        acc[mf][nf] = wmma_bf16(ah[mf], bl[nf], acc[mf][nf]);  // hi*lo
        acc[mf][nf] = wmma_bf16(al[mf], bh[nf], acc[mf][nf]);  // lo*hi
      }

    wait_async0();       // next buffer's DMA done (ran during WMMA)
    __syncthreads();     // all waves done reading current buffer
  }

  // Store transposed + split: T^t[n][m] so pass2 streams it with no transpose
  uint16_t* ThS = Th + (size_t)s * NDIM * NDIM;
  uint16_t* TlS = Tl + (size_t)s * NDIM * NDIM;
#pragma unroll
  for (int mf = 0; mf < MF; ++mf)
#pragma unroll
    for (int nf = 0; nf < NF; ++nf) {
      int ng = bn + wn + nf * 16 + l15;
      int mg = bm + wm + mf * 16 + 8 * half;
      __align__(16) uint16_t ph[8], pl[8];
#pragma unroll
      for (int j = 0; j < 8; ++j) split_bf16(acc[mf][nf][j], ph[j], pl[j]);
      *(uint4*)(ThS + (size_t)ng * NDIM + mg) = *(uint4*)ph;
      *(uint4*)(TlS + (size_t)ng * NDIM + mg) = *(uint4*)pl;
    }
}

// ---------------------------------------------------------------------------
// Pass 2: Out[s][k2][w] = sum_h C[k2][h] * T^t[s][w][h]   (fp32 out)
// Both operands pre-split in memory -> staging is pure async DMA.
// ---------------------------------------------------------------------------
__global__ __launch_bounds__(NTHREADS) void dct_pass2(
    const uint16_t* __restrict__ Gh, const uint16_t* __restrict__ Gl,
    const uint16_t* __restrict__ Th, const uint16_t* __restrict__ Tl,
    float* __restrict__ Out)
{
  __shared__ __align__(16) uint16_t sAh[2][BM * BK];
  __shared__ __align__(16) uint16_t sAl[2][BM * BK];
  __shared__ __align__(16) uint16_t sBh[2][BN * BK];
  __shared__ __align__(16) uint16_t sBl[2][BN * BK];

  const int s    = blockIdx.z;
  const int bm   = blockIdx.y * BM;
  const int bn   = blockIdx.x * BN;
  const int tid  = threadIdx.x;
  const int lane = tid & 31;
  const int wave = tid >> 5;
  const int half = lane >> 4;
  const int l15  = lane & 15;
  const int wm   = (wave >> 2) * WM;
  const int wn   = (wave & 3) * WN;

  const uint16_t* ThS = Th + (size_t)s * NDIM * NDIM;
  const uint16_t* TlS = Tl + (size_t)s * NDIM * NDIM;

  auto stage = [&](int k0, int buf) {
#pragma unroll
    for (int it = 0; it < (BM * BK / 8) / NTHREADS; ++it) {
      int i = tid + it * NTHREADS;
      int r = i >> 2;
      int c = (i & 3) << 3;
      size_t ga = (size_t)(bm + r) * NDIM + (k0 + c);
      async_copy_b128(lds_off(&sAh[buf][r * BK + c]),
                      (uint64_t)(uintptr_t)(Gh + ga));
      async_copy_b128(lds_off(&sAl[buf][r * BK + c]),
                      (uint64_t)(uintptr_t)(Gl + ga));
      size_t gb = (size_t)(bn + r) * NDIM + (k0 + c);
      async_copy_b128(lds_off(&sBh[buf][r * BK + c]),
                      (uint64_t)(uintptr_t)(ThS + gb));
      async_copy_b128(lds_off(&sBl[buf][r * BK + c]),
                      (uint64_t)(uintptr_t)(TlS + gb));
    }
  };

  v8f acc[MF][NF];
#pragma unroll
  for (int i = 0; i < MF; ++i)
#pragma unroll
    for (int j = 0; j < NF; ++j) acc[i][j] = (v8f)(0.0f);

  stage(0, 0);
  wait_async0();
  __syncthreads();

#pragma unroll 1
  for (int ki = 0; ki < KITERS; ++ki) {
    const int buf  = ki & 1;
    const int nbuf = buf ^ 1;

    if (ki + 1 < KITERS) stage((ki + 1) * BK, nbuf);

    const uint16_t* cAh = sAh[buf];
    const uint16_t* cAl = sAl[buf];
    const uint16_t* cBh = sBh[buf];
    const uint16_t* cBl = sBl[buf];

    v16bf ah[MF], al[MF], bh[NF], bl[NF];
#pragma unroll
    for (int mf = 0; mf < MF; ++mf) {
      int row = wm + mf * 16 + l15;
      ah[mf] = ldsfragA(cAh, row, half);
      al[mf] = ldsfragA(cAl, row, half);
    }
#pragma unroll
    for (int nf = 0; nf < NF; ++nf) {
      int row = wn + nf * 16 + l15;
      bh[nf] = ldsfragB(cBh, row, half);
      bl[nf] = ldsfragB(cBl, row, half);
    }
#pragma unroll
    for (int mf = 0; mf < MF; ++mf)
#pragma unroll
      for (int nf = 0; nf < NF; ++nf) {
        acc[mf][nf] = wmma_bf16(ah[mf], bh[nf], acc[mf][nf]);
        acc[mf][nf] = wmma_bf16(ah[mf], bl[nf], acc[mf][nf]);
        acc[mf][nf] = wmma_bf16(al[mf], bh[nf], acc[mf][nf]);
      }

    wait_async0();
    __syncthreads();
  }

  float* Os = Out + (size_t)s * NDIM * NDIM;
#pragma unroll
  for (int mf = 0; mf < MF; ++mf)
#pragma unroll
    for (int nf = 0; nf < NF; ++nf) {
      int ng = bn + wn + nf * 16 + l15;
      int mg = bm + wm + mf * 16 + 8 * half;
#pragma unroll
      for (int j = 0; j < 8; ++j)
        Os[(size_t)(mg + j) * NDIM + ng] = acc[mf][nf][j];
    }
}

// ---------------------------------------------------------------------------
extern "C" void kernel_launch(void* const* d_in, const int* in_sizes, int n_in,
                              void* d_out, int out_size, void* d_ws, size_t ws_size,
                              hipStream_t stream) {
  (void)in_sizes; (void)n_in; (void)out_size; (void)ws_size;
  const float* x = (const float*)d_in[0];   // [32,3,512,512] fp32
  uint8_t* ws = (uint8_t*)d_ws;

  // Workspace layout: G hi/lo (1 MB), T^t hi/lo (96 MB, L2-resident)
  uint16_t* Gh = (uint16_t*)ws;
  uint16_t* Gl = Gh + (size_t)NDIM * NDIM;
  uint16_t* Th = (uint16_t*)(ws + 2ull * NDIM * NDIM * sizeof(uint16_t));
  uint16_t* Tl = Th + (size_t)NSLICE * NDIM * NDIM;
  float* out = (float*)d_out;

  dim3 ggen(NDIM / 256, NDIM, 1);
  dct_gen_mat<<<ggen, 256, 0, stream>>>(Gh, Gl);

  dim3 gg(NDIM / BN, NDIM / BM, NSLICE);
  dct_pass1<<<gg, NTHREADS, 0, stream>>>(x, Gh, Gl, Th, Tl);
  dct_pass2<<<gg, NTHREADS, 0, stream>>>(Gh, Gl, Th, Tl, out);
}
